// ImprovedHGT_21414706938651
// MI455X (gfx1250) — matrix operand order repaired and checked by reading, not compiled, not run on hardware
//
#include <hip/hip_runtime.h>
#include <hip/hip_bf16.h>

typedef __attribute__((ext_vector_type(16))) _Float16 v16h;
typedef __attribute__((ext_vector_type(8)))  float    v8f;

#define NN_NODES 100000
#define NE_EDGES 1000000
#define F_IN 32
#define HID 64

// ---------------------------------------------------------------------------
// ordered-float <-> uint monotone encoding for atomicMax-based f32 max
// ---------------------------------------------------------------------------
__device__ __forceinline__ unsigned int enc_f(float f) {
    unsigned int u = __float_as_uint(f);
    return u ^ ((((int)u) >> 31) | 0x80000000u);
}
__device__ __forceinline__ float dec_f(unsigned int k) {
    unsigned int u = (k & 0x80000000u) ? (k ^ 0x80000000u) : ~k;
    return __uint_as_float(u);
}
#define NEG_INF_KEY 0x007FFFFFu  // enc_f(-inf)

// ---------------------------------------------------------------------------
// WMMA fragment helpers (wave32, v_wmma_f32_16x16x32_f16 layouts per ISA 7.12.2)
//   A 16x32 f16: lane l -> M=l&15 ; elems 0..7 = K (l<16?0:8)+j ; 8..15 = +16
//   B 32x16 f16: lane l -> N=l&15 ; elems 0..7 = K (l<16?0:8)+j ; 8..15 = +16
//   C 16x16 f32: lane l -> N=l&15 ; vgpr r -> M = r + (l<16?0:8)
// ---------------------------------------------------------------------------
__device__ __forceinline__ v16h load_a_frag_lds(const _Float16* A, int ldk, int lane, int k0) {
    int m  = lane & 15;
    int kA = (lane < 16) ? 0 : 8;
    v16h a{};
#pragma unroll
    for (int j = 0; j < 8; ++j) {
        a[j]     = A[m * ldk + k0 + kA + j];
        a[8 + j] = A[m * ldk + k0 + 16 + kA + j];
    }
    return a;
}
__device__ __forceinline__ v16h load_b_frag(const _Float16* W, int ldn, int lane, int k0, int n0) {
    int nB = lane & 15;
    int kB = (lane < 16) ? 0 : 8;
    v16h b{};
#pragma unroll
    for (int j = 0; j < 16; ++j)
        b[j] = W[(size_t)(k0 + kB + (j < 8 ? j : 8 + j)) * ldn + n0 + nB];
    return b;
}

// ---------------------------------------------------------------------------
// Weight preparation (single block): f16 conversions + fold relation matrices
// (a_rel into k-weights, m_rel into v-weights, incl. biases).
// ---------------------------------------------------------------------------
__global__ void prep_kernel(
    const float* enc_w_m, const float* enc_w_b,
    const float* q_w_m, const float* o_w_m, const float* q_w_b, const float* o_w_b,
    const float* k_w_m, const float* v_w_m, const float* k_w_b, const float* v_w_b,
    const float* k_b_m, const float* v_b_m, const float* k_b_b, const float* v_b_b,
    const float* a_m2b, const float* m_m2b, const float* a_b2m, const float* m_b2m,
    _Float16* enc16_m, _Float16* enc16_b,
    _Float16* qw16_m, _Float16* qw16_b, _Float16* ow16_m, _Float16* ow16_b,
    _Float16* kw16_m, _Float16* vw16_m, _Float16* kw16_b, _Float16* vw16_b,
    float* kfb_m, float* vfb_m, float* kfb_b, float* vfb_b)
{
    int t = threadIdx.x;  // 256 threads
    for (int i = t; i < F_IN * HID; i += 256) {
        enc16_m[i] = (_Float16)enc_w_m[i];
        enc16_b[i] = (_Float16)enc_w_b[i];
    }
    for (int i = t; i < HID * HID; i += 256) {
        qw16_m[i] = (_Float16)q_w_m[i];
        qw16_b[i] = (_Float16)q_w_b[i];
        ow16_m[i] = (_Float16)o_w_m[i];
        ow16_b[i] = (_Float16)o_w_b[i];
    }
    // fold: Wf[c, h*32+e] = sum_d W[c, h*32+d] * R[h, d, e]
    const float* Wsrc[4] = {k_w_m, v_w_m, k_w_b, v_w_b};
    const float* Rrel[4] = {a_m2b, m_m2b, a_b2m, m_b2m};
    _Float16*    Wdst[4] = {kw16_m, vw16_m, kw16_b, vw16_b};
    const float* Bsrc[4] = {k_b_m, v_b_m, k_b_b, v_b_b};
    float*       Bdst[4] = {kfb_m, vfb_m, kfb_b, vfb_b};
#pragma unroll
    for (int c4 = 0; c4 < 4; ++c4) {
        const float* W = Wsrc[c4]; const float* R = Rrel[c4];
        for (int i = t; i < HID * HID; i += 256) {
            int cidx = i >> 6, o = i & 63;
            int hh = o >> 5, e2 = o & 31;
            float acc = 0.f;
            for (int d2 = 0; d2 < 32; ++d2)
                acc += W[cidx * HID + hh * 32 + d2] * R[hh * 1024 + d2 * 32 + e2];
            Wdst[c4][i] = (_Float16)acc;
        }
        if (t < HID) {
            int hh = t >> 5, e2 = t & 31;
            float acc = 0.f;
            for (int d2 = 0; d2 < 32; ++d2)
                acc += Bsrc[c4][hh * 32 + d2] * R[hh * 1024 + d2 * 32 + e2];
            Bdst[c4][t] = acc;
        }
    }
}

// ---------------------------------------------------------------------------
// Fused encoder + projections. 16-node tile, 128 threads (4 waves).
//   h = relu(LN(x @ enc_w + b))  -> global (needed for residual) + LDS f16
//   q/kfold/vfold = h @ W + b    -> 6 more WMMAs per wave
// ---------------------------------------------------------------------------
__global__ __launch_bounds__(128)
void enc_proj_kernel(const float* __restrict__ x, const _Float16* __restrict__ ew16,
                     const float* __restrict__ eb, const float* __restrict__ eg,
                     const float* __restrict__ ebe,
                     const _Float16* __restrict__ qw, const _Float16* __restrict__ kw,
                     const _Float16* __restrict__ vw,
                     const float* __restrict__ qb, const float* __restrict__ kb,
                     const float* __restrict__ vb,
                     float* __restrict__ hout, float* __restrict__ qo,
                     float* __restrict__ ko, float* __restrict__ vo, int n)
{
    __shared__ _Float16 Xs[16 * F_IN];
    __shared__ float    Cs[16 * HID];
    __shared__ _Float16 Hs[16 * HID];
    __shared__ float    mu_s[16], rs_s[16];
    int t = threadIdx.x, node0 = blockIdx.x * 16;
    // stage x tile (16x32) as f16
    for (int i = t; i < 16 * F_IN; i += 128) {
        int r = i >> 5, c = i & 31;
        float v = (node0 + r < n) ? x[(size_t)(node0 + r) * F_IN + c] : 0.f;
        Xs[i] = (_Float16)v;
    }
    __syncthreads();
    int lane = t & 31, wave = t >> 5, n0 = wave * 16, nB = lane & 15;
    int mBase = (lane < 16) ? 0 : 8;
    // ---- encoder GEMM: 16x32 @ 32x64 (1 WMMA per wave) ----
    {
        v16h a = load_a_frag_lds(Xs, F_IN, lane, 0);
        v16h b = load_b_frag(ew16, HID, lane, 0, n0);
        v8f c;
        float bv = eb[n0 + nB];
#pragma unroll
        for (int r = 0; r < 8; ++r) c[r] = bv;
        c = __builtin_amdgcn_wmma_f32_16x16x32_f16(false, a, false, b, (short)0, c, false, false);
#pragma unroll
        for (int r = 0; r < 8; ++r) Cs[(mBase + r) * HID + n0 + nB] = c[r];
    }
    __syncthreads();
    // ---- LayerNorm stats ----
    if (t < 16) {
        float mu = 0.f;
        for (int d2 = 0; d2 < HID; ++d2) mu += Cs[t * HID + d2];
        mu *= (1.f / HID);
        float var = 0.f;
        for (int d2 = 0; d2 < HID; ++d2) { float dd = Cs[t * HID + d2] - mu; var += dd * dd; }
        var *= (1.f / HID);
        mu_s[t] = mu; rs_s[t] = rsqrtf(var + 1e-5f);
    }
    __syncthreads();
    // ---- h = relu(LN(.)) : write to global AND keep f16 copy in LDS ----
    for (int i = t; i < 16 * HID; i += 128) {
        int r = i >> 6, d2 = i & 63;
        float v = fmaxf((Cs[i] - mu_s[r]) * rs_s[r] * eg[d2] + ebe[d2], 0.f);
        Hs[i] = (_Float16)v;
        if (node0 + r < n) hout[(size_t)(node0 + r) * HID + d2] = v;
    }
    __syncthreads();
    // ---- projections: q, folded-k, folded-v (2 WMMAs each per wave) ----
    v16h a0 = load_a_frag_lds(Hs, HID, lane, 0);
    v16h a1 = load_a_frag_lds(Hs, HID, lane, 32);
    const _Float16* Ws[3] = {qw, kw, vw};
    const float*    Bs[3] = {qb, kb, vb};
    float*          Os[3] = {qo, ko, vo};
    for (int mm = 0; mm < 3; ++mm) {
        v16h b0 = load_b_frag(Ws[mm], HID, lane, 0, n0);
        v16h b1 = load_b_frag(Ws[mm], HID, lane, 32, n0);
        v8f c;
        float bv = Bs[mm][n0 + nB];
#pragma unroll
        for (int r = 0; r < 8; ++r) c[r] = bv;
        c = __builtin_amdgcn_wmma_f32_16x16x32_f16(false, a0, false, b0, (short)0, c, false, false);
        c = __builtin_amdgcn_wmma_f32_16x16x32_f16(false, a1, false, b1, (short)0, c, false, false);
#pragma unroll
        for (int r = 0; r < 8; ++r) {
            int row = node0 + mBase + r;
            if (row < n) Os[mm][(size_t)row * HID + n0 + nB] = c[r];
        }
    }
}

// ---------------------------------------------------------------------------
// Buffer fills
// ---------------------------------------------------------------------------
__global__ void fill_f32(float* p, float v, long n) {
    long i = (long)blockIdx.x * blockDim.x + threadIdx.x;
    long st = (long)gridDim.x * blockDim.x;
    for (; i < n; i += st) p[i] = v;
}
__global__ void fill_u32(unsigned int* p, unsigned int v, long n) {
    long i = (long)blockIdx.x * blockDim.x + threadIdx.x;
    long st = (long)gridDim.x * blockDim.x;
    for (; i < n; i += st) p[i] = v;
}

// ---------------------------------------------------------------------------
// Edge pass 1: logit = (q[dst].k'[src]) * p / sqrt(D); cache logit to global,
// atomic-max per (dst,head).  float4 gathers (rows are 128B aligned per head).
// ---------------------------------------------------------------------------
__global__ void edge_logit_max_kernel(const int* __restrict__ ei,
                                      const float* __restrict__ qdst,
                                      const float* __restrict__ ksrc,
                                      const float* __restrict__ p,
                                      float* __restrict__ lg,
                                      unsigned int* __restrict__ maxb, int nE)
{
    int i = blockIdx.x * blockDim.x + threadIdx.x;
    if (i >= nE * 2) return;
    int e = i >> 1, hh = i & 1;
    int src = ei[e], dst = ei[nE + e];
    const float4* q4 = (const float4*)(qdst + (size_t)dst * HID + hh * 32);
    const float4* k4 = (const float4*)(ksrc + (size_t)src * HID + hh * 32);
    float dot = 0.f;
#pragma unroll
    for (int d = 0; d < 8; ++d) {
        float4 qa = q4[d], ka = k4[d];
        dot += qa.x * ka.x + qa.y * ka.y + qa.z * ka.z + qa.w * ka.w;
    }
    float logit = dot * p[hh] * 0.17677669529663687f;  // 1/sqrt(32)
    lg[i] = logit;
    atomicMax(&maxb[(size_t)dst * 2 + hh], enc_f(logit));
}

// ---------------------------------------------------------------------------
// Edge pass 2: w = exp(logit - max); atomicAdd denominator + weighted v scatter.
// Reads cached logit (8B) instead of re-gathering q+k (512B).
// ---------------------------------------------------------------------------
__global__ void edge_acc_kernel(const int* __restrict__ ei,
                                const float* __restrict__ vsrc,
                                const float* __restrict__ lg,
                                const unsigned int* __restrict__ maxb,
                                float* __restrict__ sumb, float* __restrict__ agg, int nE)
{
    int i = blockIdx.x * blockDim.x + threadIdx.x;
    if (i >= nE * 2) return;
    int e = i >> 1, hh = i & 1;
    int src = ei[e], dst = ei[nE + e];
    const float* vrow = vsrc + (size_t)src * HID + hh * 32;
    __builtin_prefetch(vrow, 0, 3);  // global_prefetch_b8: pull v row while logit/max land
    float logit = lg[i];
    float m = dec_f(maxb[(size_t)dst * 2 + hh]);
    float w = expf(logit - m);
    atomicAdd(&sumb[(size_t)dst * 2 + hh], w);
    float* arow = agg + (size_t)dst * HID + hh * 32;
    const float4* v4 = (const float4*)vrow;
#pragma unroll
    for (int d = 0; d < 8; ++d) {
        float4 va = v4[d];
        atomicAdd(&arow[4 * d + 0], w * va.x);
        atomicAdd(&arow[4 * d + 1], w * va.y);
        atomicAdd(&arow[4 * d + 2], w * va.z);
        atomicAdd(&arow[4 * d + 3], w * va.w);
    }
}

// ---------------------------------------------------------------------------
// Output: a = gelu(agg/sum) @ o_w + o_b ; conv = g*a+(1-g)*h ; y = LN(conv+h)
// ---------------------------------------------------------------------------
__global__ __launch_bounds__(128)
void out_kernel(const float* __restrict__ agg, const float* __restrict__ sumb,
                const float* __restrict__ h, const _Float16* __restrict__ ow,
                const float* __restrict__ ob, const float* __restrict__ skip,
                const float* __restrict__ lng, const float* __restrict__ lnb,
                float* __restrict__ y, int n)
{
    __shared__ _Float16 As[16 * HID];
    __shared__ float    Cs[16 * HID];
    __shared__ float    mu_s[16], rs_s[16];
    int t = threadIdx.x, node0 = blockIdx.x * 16;
    for (int i = t; i < 16 * HID; i += 128) {
        int r = i >> 6, c = i & 63;
        float v = 0.f;
        if (node0 + r < n) {
            float s = sumb[(size_t)(node0 + r) * 2 + (c >> 5)];
            float x = agg[(size_t)(node0 + r) * HID + c] / (s + 1e-16f);
            v = 0.5f * x * (1.f + erff(x * 0.70710678118654752f));  // exact gelu
        }
        As[i] = (_Float16)v;
    }
    __syncthreads();
    int lane = t & 31, wave = t >> 5, n0 = wave * 16, nB = lane & 15;
    v16h a0 = load_a_frag_lds(As, HID, lane, 0);
    v16h a1 = load_a_frag_lds(As, HID, lane, 32);
    v16h b0 = load_b_frag(ow, HID, lane, 0, n0);
    v16h b1 = load_b_frag(ow, HID, lane, 32, n0);
    v8f c;
    float bv = ob[n0 + nB];
#pragma unroll
    for (int r = 0; r < 8; ++r) c[r] = bv;
    c = __builtin_amdgcn_wmma_f32_16x16x32_f16(false, a0, false, b0, (short)0, c, false, false);
    c = __builtin_amdgcn_wmma_f32_16x16x32_f16(false, a1, false, b1, (short)0, c, false, false);
    int mBase = (lane < 16) ? 0 : 8;
#pragma unroll
    for (int r = 0; r < 8; ++r) Cs[(mBase + r) * HID + n0 + nB] = c[r];
    __syncthreads();
    float gs = 1.f / (1.f + expf(-skip[0]));
    for (int i = t; i < 16 * HID; i += 128) {
        int r = i >> 6, cc = i & 63;
        if (node0 + r < n) {
            float hv = h[(size_t)(node0 + r) * HID + cc];
            Cs[i] = gs * Cs[i] + (1.f - gs) * hv + hv;  // conv + residual h
        }
    }
    __syncthreads();
    if (t < 16) {
        float mu = 0.f;
        for (int d2 = 0; d2 < HID; ++d2) mu += Cs[t * HID + d2];
        mu *= (1.f / HID);
        float var = 0.f;
        for (int d2 = 0; d2 < HID; ++d2) { float dd = Cs[t * HID + d2] - mu; var += dd * dd; }
        var *= (1.f / HID);
        mu_s[t] = mu; rs_s[t] = rsqrtf(var + 1e-5f);
    }
    __syncthreads();
    for (int i = t; i < 16 * HID; i += 128) {
        int r = i >> 6, d2 = i & 63;
        if (node0 + r < n)
            y[(size_t)(node0 + r) * HID + d2] =
                (Cs[i] - mu_s[r]) * rs_s[r] * lng[d2] + lnb[d2];
    }
}

// ---------------------------------------------------------------------------
extern "C" void kernel_launch(void* const* d_in, const int* in_sizes, int n_in,
                              void* d_out, int out_size, void* d_ws, size_t ws_size,
                              hipStream_t stream) {
    const int N = NN_NODES, E = NE_EDGES;
    // inputs in setup_inputs() dict order
    const float* x_mac   = (const float*)d_in[0];
    const float* x_buf   = (const float*)d_in[1];
    const int*   ei_m2b  = (const int*)d_in[2];
    const int*   ei_b2m  = (const int*)d_in[3];
    const float* enc_w_m = (const float*)d_in[4];
    const float* enc_b_m = (const float*)d_in[5];
    const float* enc_g_m = (const float*)d_in[6];
    const float* enc_be_m= (const float*)d_in[7];
    const float* k_w_m   = (const float*)d_in[8];
    const float* k_b_m   = (const float*)d_in[9];
    const float* q_w_m   = (const float*)d_in[10];
    const float* q_b_m   = (const float*)d_in[11];
    const float* v_w_m   = (const float*)d_in[12];
    const float* v_b_m   = (const float*)d_in[13];
    const float* o_w_m   = (const float*)d_in[14];
    const float* o_b_m   = (const float*)d_in[15];
    const float* skip_m  = (const float*)d_in[16];
    const float* enc_w_b = (const float*)d_in[17];
    const float* enc_b_b = (const float*)d_in[18];
    const float* enc_g_b = (const float*)d_in[19];
    const float* enc_be_b= (const float*)d_in[20];
    const float* k_w_b   = (const float*)d_in[21];
    const float* k_b_b   = (const float*)d_in[22];
    const float* q_w_b   = (const float*)d_in[23];
    const float* q_b_b   = (const float*)d_in[24];
    const float* v_w_b   = (const float*)d_in[25];
    const float* v_b_b   = (const float*)d_in[26];
    const float* o_w_b   = (const float*)d_in[27];
    const float* o_b_b   = (const float*)d_in[28];
    const float* skip_b  = (const float*)d_in[29];
    const float* a_m2b   = (const float*)d_in[30];
    const float* m_m2b   = (const float*)d_in[31];
    const float* p_m2b   = (const float*)d_in[32];
    const float* a_b2m   = (const float*)d_in[33];
    const float* m_b2m   = (const float*)d_in[34];
    const float* p_b2m   = (const float*)d_in[35];
    const float* ln_g    = (const float*)d_in[36];
    const float* ln_b    = (const float*)d_in[37];

    float* y_mac = (float*)d_out;
    float* y_buf = (float*)d_out + (size_t)N * HID;

    // workspace carve-up
    char* ws = (char*)d_ws;
    const size_t NH = (size_t)N * HID;
    auto takeF = [&](size_t nElem) { float* p = (float*)ws; ws += nElem * 4; return p; };
    auto takeU = [&](size_t nElem) { unsigned int* p = (unsigned int*)ws; ws += nElem * 4; return p; };
    auto takeH = [&](size_t nElem) { _Float16* p = (_Float16*)ws; ws += nElem * 2; return p; };
    float* h_m  = takeF(NH);  float* h_b  = takeF(NH);
    float* q_m  = takeF(NH);  float* kf_m = takeF(NH);  float* vf_m = takeF(NH);
    float* q_b  = takeF(NH);  float* kf_b = takeF(NH);  float* vf_b = takeF(NH);
    float* ag_m = takeF(NH);  float* ag_b = takeF(NH);
    float* lg_m2b = takeF((size_t)E * 2);
    float* lg_b2m = takeF((size_t)E * 2);
    unsigned int* mx_m = takeU((size_t)N * 2);
    unsigned int* mx_b = takeU((size_t)N * 2);
    float* sm_m = takeF((size_t)N * 2);
    float* sm_b = takeF((size_t)N * 2);
    _Float16* enc16_m = takeH(F_IN * HID); _Float16* enc16_b = takeH(F_IN * HID);
    _Float16* qw16_m = takeH(HID * HID);   _Float16* qw16_b = takeH(HID * HID);
    _Float16* ow16_m = takeH(HID * HID);   _Float16* ow16_b = takeH(HID * HID);
    _Float16* kw16_m = takeH(HID * HID);   _Float16* vw16_m = takeH(HID * HID);
    _Float16* kw16_b = takeH(HID * HID);   _Float16* vw16_b = takeH(HID * HID);
    float* kfb_m = takeF(HID); float* vfb_m = takeF(HID);
    float* kfb_b = takeF(HID); float* vfb_b = takeF(HID);

    // 1) weight prep (fold relation transforms, f16 conversion)
    prep_kernel<<<1, 256, 0, stream>>>(
        enc_w_m, enc_w_b, q_w_m, o_w_m, q_w_b, o_w_b,
        k_w_m, v_w_m, k_w_b, v_w_b, k_b_m, v_b_m, k_b_b, v_b_b,
        a_m2b, m_m2b, a_b2m, m_b2m,
        enc16_m, enc16_b, qw16_m, qw16_b, ow16_m, ow16_b,
        kw16_m, vw16_m, kw16_b, vw16_b, kfb_m, vfb_m, kfb_b, vfb_b);

    int nodeBlocks = (N + 15) / 16;
    // 2) fused encoder + q/kfold/vfold projections (7 WMMA / wave)
    enc_proj_kernel<<<nodeBlocks, 128, 0, stream>>>(
        x_mac, enc16_m, enc_b_m, enc_g_m, enc_be_m,
        qw16_m, kw16_m, vw16_m, q_b_m, kfb_m, vfb_m,
        h_m, q_m, kf_m, vf_m, N);
    enc_proj_kernel<<<nodeBlocks, 128, 0, stream>>>(
        x_buf, enc16_b, enc_b_b, enc_g_b, enc_be_b,
        qw16_b, kw16_b, vw16_b, q_b_b, kfb_b, vfb_b,
        h_b, q_b, kf_b, vf_b, N);
    // 3) init softmax stats / accumulators (contiguous buffers)
    fill_f32<<<2048, 256, 0, stream>>>(ag_m, 0.f, (long)NH * 2);
    fill_f32<<<256, 256, 0, stream>>>(sm_m, 0.f, (long)N * 4);
    fill_u32<<<256, 256, 0, stream>>>(mx_m, NEG_INF_KEY, (long)N * 4);
    // 4) edge pass 1: logits + segment max   (m2b: src=mac, dst=buf)
    int edgeBlocks = (E * 2 + 255) / 256;
    edge_logit_max_kernel<<<edgeBlocks, 256, 0, stream>>>(ei_m2b, q_b, kf_m, p_m2b, lg_m2b, mx_b, E);
    edge_logit_max_kernel<<<edgeBlocks, 256, 0, stream>>>(ei_b2m, q_m, kf_b, p_b2m, lg_b2m, mx_m, E);
    // 5) edge pass 2: exp + weighted scatter (reads cached logits)
    edge_acc_kernel<<<edgeBlocks, 256, 0, stream>>>(ei_m2b, vf_m, lg_m2b, mx_b, sm_b, ag_b, E);
    edge_acc_kernel<<<edgeBlocks, 256, 0, stream>>>(ei_b2m, vf_b, lg_b2m, mx_m, sm_m, ag_m, E);
    // 6) output projection + skip gate + residual + shared LN
    out_kernel<<<nodeBlocks, 128, 0, stream>>>(ag_m, sm_m, h_m, ow16_m, o_b_m, skip_m, ln_g, ln_b, y_mac, N);
    out_kernel<<<nodeBlocks, 128, 0, stream>>>(ag_b, sm_b, h_b, ow16_b, o_b_b, skip_b, ln_g, ln_b, y_buf, N);
}
